// GINEncoder_25933012533384
// MI455X (gfx1250) — matrix operand order repaired and checked by reading, not compiled, not run on hardware
//
#include <hip/hip_runtime.h>
#include <hip/hip_bf16.h>
#include <stdint.h>

// Problem constants (from reference)
#define NNODES 50000
#define NEDGES 800000
#define NGRAPH 8
#define D_IN   128
#define D_HID  256
#define D_OUT  128
#define BN_EPS 1e-5f

typedef __attribute__((ext_vector_type(16))) __bf16          v16bf;
typedef __attribute__((ext_vector_type(8)))  float           v8f;
typedef __attribute__((ext_vector_type(8)))  unsigned short  u16x8;
typedef __attribute__((ext_vector_type(16))) unsigned short  u16x16;

union AFrag { u16x8  h[2]; v16bf v; };
union BFrag { u16x16 u;    v16bf v; };

__device__ __forceinline__ unsigned short f32_to_bf16(float f) {
    union { float f; unsigned u; } v; v.f = f;
    unsigned u = v.u;
    u += 0x7FFFu + ((u >> 16) & 1u);   // round-to-nearest-even
    return (unsigned short)(u >> 16);
}

// ---------------------------------------------------------------- elementwise
__global__ void cvt_f32_bf16_kernel(const float* __restrict__ in,
                                    unsigned short* __restrict__ out,
                                    long long n) {
    for (long long i = blockIdx.x * (long long)blockDim.x + threadIdx.x;
         i < n; i += (long long)gridDim.x * blockDim.x)
        out[i] = f32_to_bf16(in[i]);
}

// z = (1 + eps) * h   (init of aggregation buffer; eps is a device scalar)
__global__ void init_z_kernel(const float* __restrict__ h,
                              const float* __restrict__ eps,
                              float* __restrict__ z, long long n) {
    const float s = 1.0f + eps[0];
    for (long long i = blockIdx.x * (long long)blockDim.x + threadIdx.x;
         i < n; i += (long long)gridDim.x * blockDim.x)
        z[i] = s * h[i];
}

// ---------------------------------------------------------------- aggregation
// One thread per (edge, 4-float chunk): z[dst] += h[src]
// h (<= 51 MB) is L2-resident on the 192 MB L2, so scattered gathers/atomics
// stay on-chip.
__global__ void scatter_add_kernel(const float* __restrict__ h,
                                   const int* __restrict__ src,
                                   const int* __restrict__ dst,
                                   float* __restrict__ z,
                                   int D, int cl2, long long total) {
    const int cmask = (1 << cl2) - 1;
    for (long long i = blockIdx.x * (long long)blockDim.x + threadIdx.x;
         i < total; i += (long long)gridDim.x * blockDim.x) {
        const int e = (int)(i >> cl2);
        const int c = (int)(i & cmask) * 4;
        const int s = src[e], d = dst[e];
        const float4 v = *(const float4*)(h + (size_t)s * D + c);
        float* zp = z + (size_t)d * D + c;
        atomicAdd(zp + 0, v.x);
        atomicAdd(zp + 1, v.y);
        atomicAdd(zp + 2, v.z);
        atomicAdd(zp + 3, v.w);
    }
}

// ---------------------------------------------------------------- WMMA GEMM
// out[M x Nout] = A[M x K](bf16) * W[Nout x K](bf16)^T + bias, opt. ReLU.
// The full weight matrix (<=128 KB) is staged into LDS once per block
// (320 KB/WGP on CDNA5); B fragments then come from ds_load_b128 so the
// inner loop issues only the 2 per-wave A loads to vmem per K-step.
// One wave computes a 16x64 output strip; 8 waves / 256-thread block.
__global__ __launch_bounds__(256)
void gemm_bias_act_kernel(const unsigned short* __restrict__ A,
                          const unsigned short* __restrict__ W,
                          const float* __restrict__ bias,
                          float* __restrict__ outF,
                          unsigned short* __restrict__ outB,
                          int M, int K, int Nout, int relu, int out_bf16) {
    extern __shared__ unsigned short sW[];   // Nout*K bf16 elements

    // Cooperative 16B-vector copy of the whole weight matrix into LDS.
    {
        const int chunks = (Nout * K) >> 3;          // u16x8 chunks
        const u16x8* __restrict__ gsrc = (const u16x8*)W;
        u16x8* __restrict__ ldst = (u16x8*)sW;
        for (int idx = threadIdx.x; idx < chunks; idx += 256)
            ldst[idx] = gsrc[idx];
    }
    __syncthreads();

    const int lane    = threadIdx.x & 31;
    const int wave    = threadIdx.x >> 5;
    const int tiles_n = Nout >> 6;                 // 64-wide strips
    const int tiles_m = M >> 4;
    const int tile    = blockIdx.x * 8 + wave;
    if (tile >= tiles_m * tiles_n) return;         // wave-uniform exit (after barrier)
    const int nt    = tile % tiles_n;
    const int mt    = tile / tiles_n;
    const int r     = lane & 15;
    const int hsel  = lane >> 4;
    const int nbase = nt * 64;

    const unsigned short* aptr = A + (size_t)(mt * 16 + r) * K;

    v8f c0 = {}, c1 = {}, c2 = {}, c3 = {};

    for (int kb = 0; kb < K; kb += 32) {
        AFrag a;
        a.h[0] = *(const u16x8*)(aptr + kb + hsel * 16);
        a.h[1] = *(const u16x8*)(aptr + kb + hsel * 16 + 8);
        const int wk = kb + hsel * 16;
        BFrag b0, b1, b2, b3;
        b0.u = *(const u16x16*)(sW + (nbase +  0 + r) * K + wk);
        b1.u = *(const u16x16*)(sW + (nbase + 16 + r) * K + wk);
        b2.u = *(const u16x16*)(sW + (nbase + 32 + r) * K + wk);
        b3.u = *(const u16x16*)(sW + (nbase + 48 + r) * K + wk);
        c0 = __builtin_amdgcn_wmma_f32_16x16x32_bf16(false, a.v, false, b0.v, (short)0, c0, false, false);
        c1 = __builtin_amdgcn_wmma_f32_16x16x32_bf16(false, a.v, false, b1.v, (short)0, c1, false, false);
        c2 = __builtin_amdgcn_wmma_f32_16x16x32_bf16(false, a.v, false, b2.v, (short)0, c2, false, false);
        c3 = __builtin_amdgcn_wmma_f32_16x16x32_bf16(false, a.v, false, b3.v, (short)0, c3, false, false);
    }

    const int mrow0 = mt * 16 + hsel * 8;   // D-matrix: m = (lane>>4)*8 + j, n = lane&15
#define STORE_ACC(cc, nn)                                                     \
    do {                                                                      \
        const int ncol = nbase + (nn) + r;                                    \
        const float bv = bias[ncol];                                          \
        _Pragma("unroll")                                                     \
        for (int j = 0; j < 8; ++j) {                                         \
            float v = cc[j] + bv;                                             \
            if (relu) v = fmaxf(v, 0.0f);                                     \
            const size_t off = (size_t)(mrow0 + j) * Nout + ncol;             \
            if (out_bf16) outB[off] = f32_to_bf16(v);                         \
            else          outF[off] = v;                                      \
        }                                                                     \
    } while (0)
    STORE_ACC(c0,  0);
    STORE_ACC(c1, 16);
    STORE_ACC(c2, 32);
    STORE_ACC(c3, 48);
#undef STORE_ACC
}

// ---------------------------------------------------------------- batch norm
// Per-channel partial sums per block, atomics into stats[0..D) = sum,
// stats[D..2D) = sum of squares.
__global__ void bn_stats_kernel(const float* __restrict__ z,
                                float* __restrict__ stats, int Nn, int D) {
    const int c = threadIdx.x;
    if (c >= D) return;
    float s = 0.0f, ss = 0.0f;
    for (int rr = blockIdx.x; rr < Nn; rr += gridDim.x) {
        const float v = z[(size_t)rr * D + c];
        s += v; ss += v * v;
    }
    atomicAdd(&stats[c], s);
    atomicAdd(&stats[D + c], ss);
}

__global__ void bn_apply_kernel(const float* __restrict__ z,
                                const float* __restrict__ stats,
                                const float* __restrict__ gamma,
                                const float* __restrict__ beta,
                                float* __restrict__ out,
                                long long total, int D, float invN, int relu) {
    for (long long i = blockIdx.x * (long long)blockDim.x + threadIdx.x;
         i < total; i += (long long)gridDim.x * blockDim.x) {
        const int c  = (int)(i & (D - 1));
        const float mu  = stats[c] * invN;
        const float var = stats[D + c] * invN - mu * mu;
        float v = (z[i] - mu) * rsqrtf(var + BN_EPS) * gamma[c] + beta[c];
        if (relu) v = fmaxf(v, 0.0f);
        out[i] = v;
    }
}

// ---------------------------------------------------------------- readout
__global__ void pool_sum_kernel(const float* __restrict__ h,
                                const int* __restrict__ gid,
                                float* __restrict__ sums,
                                float* __restrict__ counts,
                                long long total /* N*D_OUT */) {
    for (long long i = blockIdx.x * (long long)blockDim.x + threadIdx.x;
         i < total; i += (long long)gridDim.x * blockDim.x) {
        const int n = (int)(i >> 7);          // D_OUT == 128
        const int c = (int)(i & 127);
        const int g = gid[n];
        atomicAdd(&sums[(size_t)g * D_OUT + c], h[i]);
        if (c == 0) atomicAdd(&counts[g], 1.0f);
    }
}

__global__ void pool_div_kernel(const float* __restrict__ sums,
                                const float* __restrict__ counts,
                                float* __restrict__ out, int total) {
    const int i = blockIdx.x * blockDim.x + threadIdx.x;
    if (i < total) {
        const int g = i / D_OUT;
        out[i] = sums[i] / fmaxf(counts[g], 1.0f);
    }
}

// ---------------------------------------------------------------- host
static inline int gs_blocks(long long n) {
    long long b = (n + 255) / 256;
    return (int)(b > 8192 ? 8192 : (b < 1 ? 1 : b));
}

extern "C" void kernel_launch(void* const* d_in, const int* in_sizes, int n_in,
                              void* d_out, int out_size, void* d_ws, size_t ws_size,
                              hipStream_t stream) {
    (void)in_sizes; (void)n_in; (void)out_size; (void)ws_size;

    const float* x   = (const float*)d_in[0];
    const int*   src = (const int*)d_in[1];
    const int*   dst = (const int*)d_in[2];
    const int*   gid = (const int*)d_in[3];

    // Workspace carve (256B aligned)
    char*  ws  = (char*)d_ws;
    size_t off = 0;
    auto carve = [&](size_t bytes) -> void* {
        void* p = ws + off;
        off += (bytes + 255) & ~(size_t)255;
        return p;
    };
    float*          zbuf  = (float*)carve((size_t)NNODES * 256 * 4); // agg/z, then z2
    float*          hbuf  = (float*)carve((size_t)NNODES * 256 * 4); // layer output
    unsigned short* zb    = (unsigned short*)carve((size_t)NNODES * 256 * 2);
    unsigned short* ab    = (unsigned short*)carve((size_t)NNODES * 256 * 2);
    unsigned short* wb1   = (unsigned short*)carve((size_t)256 * 256 * 2);
    unsigned short* wb2   = (unsigned short*)carve((size_t)256 * 256 * 2);
    float*          stats = (float*)carve(2 * 256 * 4);
    float*          psum  = (float*)carve((size_t)NGRAPH * D_OUT * 4);
    float*          pcnt  = (float*)carve((size_t)NGRAPH * 4);

    const float* h = x;   // current layer input (fp32)

    for (int i = 0; i < 3; ++i) {
        const int Din  = (i == 0) ? D_IN : D_HID;
        const int Dmid = D_HID;
        const int Dout = (i == 2) ? D_OUT : D_HID;

        const float* w1  = (const float*)d_in[4 + 7 * i + 0];
        const float* b1  = (const float*)d_in[4 + 7 * i + 1];
        const float* w2  = (const float*)d_in[4 + 7 * i + 2];
        const float* b2  = (const float*)d_in[4 + 7 * i + 3];
        const float* eps = (const float*)d_in[4 + 7 * i + 4];
        const float* gam = (const float*)d_in[4 + 7 * i + 5];
        const float* bet = (const float*)d_in[4 + 7 * i + 6];

        // bf16 weights
        cvt_f32_bf16_kernel<<<gs_blocks((long long)Dmid * Din), 256, 0, stream>>>(
            w1, wb1, (long long)Dmid * Din);
        cvt_f32_bf16_kernel<<<gs_blocks((long long)Dout * Dmid), 256, 0, stream>>>(
            w2, wb2, (long long)Dout * Dmid);

        // z = (1+eps)*h, then z[dst] += h[src] over all edges
        const long long nz = (long long)NNODES * Din;
        init_z_kernel<<<gs_blocks(nz), 256, 0, stream>>>(h, eps, zbuf, nz);
        const int cl2 = (Din == 128) ? 5 : 6;            // chunks of 4 floats
        const long long et = (long long)NEDGES << cl2;
        scatter_add_kernel<<<8192, 256, 0, stream>>>(h, src, dst, zbuf, Din, cl2, et);

        // z -> bf16
        cvt_f32_bf16_kernel<<<gs_blocks(nz), 256, 0, stream>>>(zbuf, zb, nz);

        // GEMM1: a = relu(z @ w1^T + b1)  (bf16 out)
        {
            const int tiles = (NNODES / 16) * (Dmid / 64);
            const size_t shmem = (size_t)Dmid * Din * sizeof(unsigned short);
            gemm_bias_act_kernel<<<(tiles + 7) / 8, 256, shmem, stream>>>(
                zb, wb1, b1, nullptr, ab, NNODES, Din, Dmid, /*relu=*/1, /*out_bf16=*/1);
        }
        // GEMM2: z2 = a @ w2^T + b2  (fp32 out, reuse zbuf)
        {
            const int tiles = (NNODES / 16) * (Dout / 64);
            const size_t shmem = (size_t)Dout * Dmid * sizeof(unsigned short);
            gemm_bias_act_kernel<<<(tiles + 7) / 8, 256, shmem, stream>>>(
                ab, wb2, b2, zbuf, nullptr, NNODES, Dmid, Dout, /*relu=*/0, /*out_bf16=*/0);
        }

        // BatchNorm (training-mode batch stats) + inter-layer ReLU
        hipMemsetAsync(stats, 0, 2 * Dout * sizeof(float), stream);
        bn_stats_kernel<<<2048, Dout, 0, stream>>>(zbuf, stats, NNODES, Dout);
        const long long no = (long long)NNODES * Dout;
        bn_apply_kernel<<<gs_blocks(no), 256, 0, stream>>>(
            zbuf, stats, gam, bet, hbuf, no, Dout, 1.0f / (float)NNODES,
            /*relu=*/(i < 2) ? 1 : 0);

        h = hbuf;
    }

    // Per-graph mean readout
    hipMemsetAsync(psum, 0, (size_t)NGRAPH * D_OUT * sizeof(float), stream);
    hipMemsetAsync(pcnt, 0, (size_t)NGRAPH * sizeof(float), stream);
    const long long pt = (long long)NNODES * D_OUT;
    pool_sum_kernel<<<gs_blocks(pt), 256, 0, stream>>>(h, gid, psum, pcnt, pt);
    pool_div_kernel<<<(NGRAPH * D_OUT + 255) / 256, 256, 0, stream>>>(
        psum, pcnt, (float*)d_out, NGRAPH * D_OUT);
}
